// CNNLSTM_59193239273595
// MI455X (gfx1250) — matrix operand order repaired and checked by reading, not compiled, not run on hardware
//
#include <hip/hip_runtime.h>
#include <hip/hip_bf16.h>
#include <cstdint>
#include <cstddef>

// ---------------- CDNA5 WMMA types ----------------
typedef __attribute__((ext_vector_type(16))) _Float16 v16h;
typedef __attribute__((ext_vector_type(8)))  float    v8f;
typedef __attribute__((ext_vector_type(4)))  unsigned int u32x4;
typedef __attribute__((ext_vector_type(8)))  int      i32x8;
typedef __attribute__((ext_vector_type(4)))  int      i32x4;

union V16H { v16h v; uint4 q[2]; };

__device__ __forceinline__ v8f wmma_f16(v16h a, v16h b, v8f c) {
  // (neg_a, A, neg_b, B, c_mod, C, reuse_a, reuse_b)
  return __builtin_amdgcn_wmma_f32_16x16x32_f16(false, a, false, b, (short)0, c, false, false);
}

#if __has_builtin(__builtin_amdgcn_tensor_load_to_lds)
#define HAVE_TDM 1
#else
#define HAVE_TDM 0
#if defined(__HIP_DEVICE_COMPILE__)
#warning "CDNA5 TDM builtin not available on device pass - using cooperative-load fallback"
#endif
#endif

// ---------------- problem constants ----------------
#define BSZ   64
#define LSEQ  4096
#define LPAD  4104      // LSEQ + 8 zero pad rows so conv A-tiles never read OOB
#define EDIM  128
#define FDIM  64
#define KW    5
#define TLEN  1023      // pooled length (4092/4)
#define TPAD  1024
#define HDIM  128
#define G4H   512

// ---------------- workspace layout (bytes) ----------------
constexpr size_t OFF_HEMB  = 0;
constexpr size_t SZ_HEMB   = (size_t)BSZ * LPAD * EDIM * 2;     // f16 embeddings
constexpr size_t OFF_HC    = OFF_HEMB + SZ_HEMB;
constexpr size_t SZ_HC     = (size_t)BSZ * TPAD * FDIM * 2;     // pooled conv out, f16
constexpr size_t OFF_XG    = OFF_HC + SZ_HC;
constexpr size_t SZ_XG     = (size_t)BSZ * TPAD * G4H * 4;      // input projections, f32
constexpr size_t OFF_WCONV = OFF_XG + SZ_XG;                    // 4*20*32*16 f16
constexpr size_t OFF_WIH   = OFF_WCONV + (size_t)4*20*32*16*2;  // 32*2*32*16 f16
constexpr size_t OFF_WHH   = OFF_WIH   + (size_t)32*2*32*16*2;  // 32*4*32*16 f16

// ============================================================
// TDM: 2-D tile load (16 rows x 512 f32, row stride TPAD*512 elems)
// into LDS at lds_off. Issued by one wave; tracked by TENSORcnt.
// D# bitfields per CDNA5 ISA ch.8 (count=1, type=2 "image",
// data_size=4B, tile_dim0=512, tile_dim1=16).
// ============================================================
#if HAVE_TDM
__device__ __forceinline__ void tdm_load_xg_tile(const float* gsrc, unsigned lds_off) {
  unsigned long long ga = (unsigned long long)(uintptr_t)gsrc;
  u32x4 g0;
  g0.x = 1u;                                        // count=1 (valid), flags=0
  g0.y = lds_off;                                   // lds_addr
  g0.z = (unsigned)(ga & 0xffffffffu);              // global_addr[31:0]
  g0.w = (unsigned)((ga >> 32) & 0x1ffffffu) | (2u << 30);  // addr[56:32] | type=2
  i32x8 g1;
  g1[0] = (2 << 16);          // workgroup_mask=0, data_size=2 (4 bytes)
  g1[1] = (512 << 16);        // tensor_dim0[15:0]=512 (abar_addr=0)
  g1[2] = (16 << 16);         // tensor_dim0[31:16]=0, tensor_dim1[15:0]=16
  g1[3] = (512 << 16);        // tensor_dim1[31:16]=0, tile_dim0=512
  g1[4] = 16;                 // tile_dim1=16, tile_dim2=0
  g1[5] = (int)(TPAD * G4H);  // tensor_dim0_stride[31:0] (elements)
  g1[6] = 0;
  g1[7] = 0;
  i32x4 z4 = {0, 0, 0, 0};
#if defined(__clang_major__) && (__clang_major__ >= 23)
  i32x8 z8 = {0, 0, 0, 0, 0, 0, 0, 0};
  __builtin_amdgcn_tensor_load_to_lds(g0, g1, z4, z4, z8, 0);
#else
  __builtin_amdgcn_tensor_load_to_lds(g0, g1, z4, z4, 0);
#endif
}
#endif

// ============================================================
// 1) Embedding gather -> f16, with zero padding rows
// ============================================================
__global__ void k_embed(const int* __restrict__ x, const float* __restrict__ emb,
                        _Float16* __restrict__ hemb) {
  int blk = blockIdx.x;               // b*LPAD + l
  int b = blk / LPAD, l = blk % LPAD;
  int e = threadIdx.x;                // 128 threads
  float v = 0.f;
  if (l < LSEQ) {
    int tok = x[(size_t)b * LSEQ + l];
    v = emb[(size_t)tok * EDIM + e];
  }
  hemb[((size_t)b * LPAD + l) * EDIM + e] = (_Float16)v;
}

// ============================================================
// 2) Weight packing into per-lane WMMA B-operand layout.
// ============================================================
__global__ void k_pack_conv(const float* __restrict__ w, _Float16* __restrict__ out) {
  int idx = blockIdx.x * 256 + threadIdx.x;  // 4*20*32*16 = 40960
  if (idx >= 4 * 20 * 32 * 16) return;
  int i    = idx & 15;
  int lane = (idx >> 4) & 31;
  int kc   = (idx >> 9) % 20;
  int ft   = idx / (16 * 32 * 20);
  int r = kc * 32 + ((lane >= 16) ? 8 : 0) + (i & 7) + ((i >> 3) << 4);
  int n = ft * 16 + (lane & 15);
  int k = r / EDIM, e = r % EDIM;
  out[idx] = (_Float16)w[(size_t)n * (EDIM * KW) + (size_t)e * KW + k];  // conv_w[F][E][K]
}

__global__ void k_pack_b(const float* __restrict__ w, _Float16* __restrict__ out,
                         int kdim, int nkc, int total) {
  int idx = blockIdx.x * 256 + threadIdx.x;
  if (idx >= total) return;
  int i    = idx & 15;
  int lane = (idx >> 4) & 31;
  int kc   = (idx >> 9) % nkc;
  int nt   = idx / (512 * nkc);
  int r  = kc * 32 + ((lane >= 16) ? 8 : 0) + (i & 7) + ((i >> 3) << 4);
  int ng = nt * 16 + (lane & 15);
  out[idx] = (_Float16)w[(size_t)ng * kdim + r];
}

// ============================================================
// 3) Conv1d as WMMA GEMM + bias + relu + maxpool(4), fused.
//    FOUR 16-position l-tiles per wave: each B tile feeds 4 WMMAs,
//    4 independent accumulation chains hide the B-load latency.
// ============================================================
__global__ void __launch_bounds__(256) k_conv(const _Float16* __restrict__ hemb,
                                              const _Float16* __restrict__ wconv,
                                              const float* __restrict__ conv_b,
                                              _Float16* __restrict__ hc) {
  int b    = blockIdx.x >> 3;          // 8 blocks per batch item
  int tch  = blockIdx.x & 7;
  int wv   = threadIdx.x >> 5;         // 8 waves
  int lane = threadIdx.x & 31;
  int l0   = ((tch * 8 + wv) * 4) * 16;  // wave owns l-tiles {l0 .. l0+48}
  int m    = lane & 15;
  int b8a  = (lane < 16) ? 0 : 8;

  v8f acc[4][4];
#pragma unroll
  for (int u = 0; u < 4; ++u)
#pragma unroll
    for (int ft = 0; ft < 4; ++ft) acc[u][ft] = (v8f){0,0,0,0,0,0,0,0};

  for (int kc = 0; kc < 20; ++kc) {
    int k  = kc >> 2;
    int e0 = (kc & 3) * 32;
    V16H a[4];
    const _Float16* rp = hemb + ((size_t)b * LPAD + (l0 + m + k)) * EDIM + e0 + b8a;
#pragma unroll
    for (int u = 0; u < 4; ++u) {
      const _Float16* rpu = rp + (size_t)(u * 16) * EDIM;
      a[u].q[0] = *(const uint4*)rpu;
      a[u].q[1] = *(const uint4*)(rpu + 16);
      __builtin_prefetch(rpu + 32, 0, 1);   // next A chunk -> global_prefetch_b8
    }
#pragma unroll
    for (int ft = 0; ft < 4; ++ft) {
      V16H bm;
      const uint4* bp = (const uint4*)(wconv + ((size_t)((ft * 20 + kc) * 32 + lane)) * 16);
      bm.q[0] = bp[0]; bm.q[1] = bp[1];
#pragma unroll
      for (int u = 0; u < 4; ++u)
        acc[u][ft] = wmma_f16(a[u].v, bm.v, acc[u][ft]);
    }
  }

#pragma unroll
  for (int u = 0; u < 4; ++u) {
    int lpbase = ((l0 + u * 16) >> 2) + ((lane < 16) ? 0 : 2);
#pragma unroll
    for (int ft = 0; ft < 4; ++ft) {
      int f = ft * 16 + m;
      float bias = conv_b[f];
      float q0 = fmaxf(fmaxf(acc[u][ft][0], acc[u][ft][1]), fmaxf(acc[u][ft][2], acc[u][ft][3]));
      float q1 = fmaxf(fmaxf(acc[u][ft][4], acc[u][ft][5]), fmaxf(acc[u][ft][6], acc[u][ft][7]));
      q0 = fmaxf(q0 + bias, 0.f);
      q1 = fmaxf(q1 + bias, 0.f);
      if (lpbase < TLEN)     hc[((size_t)b * TPAD + lpbase)     * FDIM + f] = (_Float16)q0;
      if (lpbase + 1 < TLEN) hc[((size_t)b * TPAD + lpbase + 1) * FDIM + f] = (_Float16)q1;
    }
  }
}

// ============================================================
// 4) LSTM input projection: xg = hc @ w_ih^T + (b_ih + b_hh), f32 out.
// ============================================================
__global__ void __launch_bounds__(256) k_proj(const _Float16* __restrict__ hc,
                                              const _Float16* __restrict__ wih,
                                              const float* __restrict__ b_ih,
                                              const float* __restrict__ b_hh,
                                              float* __restrict__ xg) {
  int b    = blockIdx.x >> 3;
  int tch  = blockIdx.x & 7;
  int wv   = threadIdx.x >> 5;
  int lane = threadIdx.x & 31;
  int t0   = (tch * 8 + wv) * 16;      // 64 t-tiles per batch item
  int m    = lane & 15;
  int b8a  = (lane < 16) ? 0 : 8;
  int mro  = (lane < 16) ? 0 : 8;

  V16H a0, a1;                          // A reused across all 32 N-tiles
  const _Float16* rp = hc + ((size_t)b * TPAD + t0 + m) * FDIM;
  a0.q[0] = *(const uint4*)(rp + b8a);
  a0.q[1] = *(const uint4*)(rp + b8a + 16);
  a1.q[0] = *(const uint4*)(rp + 32 + b8a);
  a1.q[1] = *(const uint4*)(rp + 32 + b8a + 16);

  for (int nt = 0; nt < 32; ++nt) {
    v8f acc = (v8f){0,0,0,0,0,0,0,0};
    V16H bm;
    const uint4* bp0 = (const uint4*)(wih + ((size_t)((nt * 2 + 0) * 32 + lane)) * 16);
    bm.q[0] = bp0[0]; bm.q[1] = bp0[1];
    acc = wmma_f16(a0.v, bm.v, acc);
    const uint4* bp1 = (const uint4*)(wih + ((size_t)((nt * 2 + 1) * 32 + lane)) * 16);
    bm.q[0] = bp1[0]; bm.q[1] = bp1[1];
    acc = wmma_f16(a1.v, bm.v, acc);

    int ng = nt * 16 + m;
    float bias = b_ih[ng] + b_hh[ng];
#pragma unroll
    for (int v = 0; v < 8; ++v) {
      int trow = t0 + v + mro;
      xg[((size_t)b * TPAD + trow) * G4H + ng] = acc[v] + bias;
    }
  }
}

// ============================================================
// 5) Sequential LSTM scan. 4 blocks x 16 batch rows.
//    - w_hh B-tiles hoisted into registers (loop invariant, 16 tiles/wave)
//    - per-step xg slice staged into double-buffered LDS by the
//      Tensor Data Mover one step ahead (overlaps the WMMA phase),
//      published with s_wait_tensorcnt + workgroup barrier.
// ============================================================
__global__ void __launch_bounds__(256) k_scan(const float* __restrict__ xg,
                                              const _Float16* __restrict__ whh,
                                              const float* __restrict__ fc_w,
                                              const float* __restrict__ fc_b,
                                              float* __restrict__ out) {
  __shared__ _Float16 hp[16][HDIM];
  __shared__ float    cp[16][HDIM];
  __shared__ float    xbuf[2][16][G4H];     // 2 x 32 KB staging
  int b0   = blockIdx.x * 16;
  int wv   = threadIdx.x >> 5;
  int lane = threadIdx.x & 31;
  int m    = lane & 15;
  int b8a  = (lane < 16) ? 0 : 8;
  int mro  = (lane < 16) ? 0 : 8;
  int jg   = wv * 16 + m;

  // Hoist recurrent-weight B tiles into registers (gi x kc = 16 tiles).
  V16H bw[16];
#pragma unroll
  for (int gi = 0; gi < 4; ++gi)
#pragma unroll
    for (int kc = 0; kc < 4; ++kc) {
      int nt = gi * 8 + wv;
      const uint4* bp = (const uint4*)(whh + ((size_t)((nt * 4 + kc) * 32 + lane)) * 16);
      bw[gi * 4 + kc].q[0] = bp[0];
      bw[gi * 4 + kc].q[1] = bp[1];
    }

  for (int i = threadIdx.x; i < 16 * HDIM; i += 256) {
    hp[i >> 7][i & 127] = (_Float16)0.f;
    cp[i >> 7][i & 127] = 0.f;
  }

  // Prefill xbuf[0] with step 0.
#if HAVE_TDM
  if (wv == 0) {
    tdm_load_xg_tile(xg + ((size_t)b0 * TPAD + 0) * G4H,
                     (unsigned)(uintptr_t)&xbuf[0][0][0]);
    __builtin_amdgcn_s_wait_tensorcnt(0);
  }
#else
  for (int i = threadIdx.x; i < 16 * G4H; i += 256)
    xbuf[0][i >> 9][i & 511] = xg[((size_t)(b0 + (i >> 9)) * TPAD + 0) * G4H + (i & 511)];
#endif
  __syncthreads();

  for (int t = 0; t < TLEN; ++t) {
    int tn = t + 1;
    // Kick DMA for step t+1 into the other buffer (overlaps WMMA below).
#if HAVE_TDM
    if (wv == 0 && tn < TLEN)
      tdm_load_xg_tile(xg + ((size_t)b0 * TPAD + tn) * G4H,
                       (unsigned)(uintptr_t)&xbuf[tn & 1][0][0]);
#else
    if (tn < TLEN)
      for (int i = threadIdx.x; i < 16 * G4H; i += 256)
        xbuf[tn & 1][i >> 9][i & 511] =
            xg[((size_t)(b0 + (i >> 9)) * TPAD + tn) * G4H + (i & 511)];
#endif

    v8f ga[4];
#pragma unroll
    for (int gi = 0; gi < 4; ++gi) {
      v8f acc = (v8f){0,0,0,0,0,0,0,0};
#pragma unroll
      for (int kc = 0; kc < 4; ++kc) {
        V16H a;
        const _Float16* rp = &hp[m][kc * 32 + b8a];
        a.q[0] = *(const uint4*)rp;
        a.q[1] = *(const uint4*)(rp + 16);
        acc = wmma_f16(a.v, bw[gi * 4 + kc].v, acc);
      }
      ga[gi] = acc;
    }

    float cn[8], hn[8];
    const float (*xrow)[G4H] = xbuf[t & 1];
#pragma unroll
    for (int v = 0; v < 8; ++v) {
      int mb = v + mro;  // batch row in this block
      float xi  = ga[0][v] + xrow[mb][0 * HDIM + jg];
      float xf  = ga[1][v] + xrow[mb][1 * HDIM + jg];
      float xgv = ga[2][v] + xrow[mb][2 * HDIM + jg];
      float xo  = ga[3][v] + xrow[mb][3 * HDIM + jg];
      float iv = 1.f / (1.f + __expf(-xi));
      float fv = 1.f / (1.f + __expf(-xf));
      float gv = tanhf(xgv);
      float ov = 1.f / (1.f + __expf(-xo));
      float c  = fv * cp[mb][jg] + iv * gv;
      cn[v] = c;
      hn[v] = ov * tanhf(c);
    }
    __syncthreads();   // all waves done reading hp/xbuf for this step
#pragma unroll
    for (int v = 0; v < 8; ++v) {
      int mb = v + mro;
      cp[mb][jg] = cn[v];
      hp[mb][jg] = (_Float16)hn[v];
    }
#if HAVE_TDM
    if (wv == 0) __builtin_amdgcn_s_wait_tensorcnt(0);  // t+1 tile landed
#endif
    __syncthreads();   // publish new h + staged xg before next step
  }

  // Final FC: [16,128] @ [128,2]^T + b
  if (threadIdx.x < 32) {
    int mm = threadIdx.x >> 1;
    int c  = threadIdx.x & 1;
    float s = 0.f;
    for (int h = 0; h < HDIM; ++h) s += (float)hp[mm][h] * fc_w[c * HDIM + h];
    out[(size_t)(b0 + mm) * 2 + c] = s + fc_b[c];
  }
}

// ============================================================
extern "C" void kernel_launch(void* const* d_in, const int* in_sizes, int n_in,
                              void* d_out, int out_size, void* d_ws, size_t ws_size,
                              hipStream_t stream) {
  (void)in_sizes; (void)n_in; (void)out_size; (void)ws_size;
  const int*   x      = (const int*)  d_in[0];
  const float* emb    = (const float*)d_in[1];
  const float* conv_w = (const float*)d_in[2];
  const float* conv_b = (const float*)d_in[3];
  const float* w_ih   = (const float*)d_in[4];
  const float* w_hh   = (const float*)d_in[5];
  const float* b_ih   = (const float*)d_in[6];
  const float* b_hh   = (const float*)d_in[7];
  const float* fc_w   = (const float*)d_in[8];
  const float* fc_b   = (const float*)d_in[9];
  float* out = (float*)d_out;

  char* ws = (char*)d_ws;
  _Float16* hemb  = (_Float16*)(ws + OFF_HEMB);
  _Float16* hc    = (_Float16*)(ws + OFF_HC);
  float*    xgb   = (float*)   (ws + OFF_XG);
  _Float16* wconv = (_Float16*)(ws + OFF_WCONV);
  _Float16* wih   = (_Float16*)(ws + OFF_WIH);
  _Float16* whh   = (_Float16*)(ws + OFF_WHH);

  k_embed<<<BSZ * LPAD, 128, 0, stream>>>(x, emb, hemb);
  k_pack_conv<<<160, 256, 0, stream>>>(conv_w, wconv);
  k_pack_b<<<128, 256, 0, stream>>>(w_ih, wih, 64, 2, 32 * 2 * 32 * 16);
  k_pack_b<<<256, 256, 0, stream>>>(w_hh, whh, 128, 4, 32 * 4 * 32 * 16);
  k_conv<<<BSZ * 8, 256, 0, stream>>>(hemb, wconv, conv_b, hc);
  k_proj<<<BSZ * 8, 256, 0, stream>>>(hc, wih, b_ih, b_hh, xgb);
  k_scan<<<4, 256, 0, stream>>>(xgb, whh, fc_w, fc_b, out);
}
